// Attn_25409026523803
// MI455X (gfx1250) — matrix operand order repaired and checked by reading, not compiled, not run on hardware
//
#include <hip/hip_runtime.h>
#include <hip/hip_bf16.h>

// ---------------------------------------------------------------------------
// CDNA5 (gfx1250) fused causal-MHA: bf16 WMMA GEMMs + flash attention.
// wave32; v_wmma_f32_16x16x32_bf16; async global->LDS (ASYNCcnt) where the
// tile needs no transform; ping-pong LDS buffers -> one barrier per K-step.
// ---------------------------------------------------------------------------

#if defined(__has_builtin)
#  if __has_builtin(__builtin_amdgcn_global_load_async_to_lds_b128)
#    define HAVE_ASYNC_LDS 1
#  endif
#endif
#ifndef HAVE_ASYNC_LDS
#  define HAVE_ASYNC_LDS 0
#endif

typedef __bf16 bf16_t;
typedef __bf16        v16bf __attribute__((ext_vector_type(16)));
typedef float         v8f   __attribute__((ext_vector_type(8)));
typedef unsigned int  u32x4 __attribute__((ext_vector_type(4)));

// gcc-style vector matching the builtin's "V4i" prototype spelling.
typedef int gvec4i __attribute__((vector_size(16)));
typedef __attribute__((address_space(1))) gvec4i* as1_v4i;
typedef __attribute__((address_space(3))) gvec4i* as3_v4i;

union Frag16 { u32x4 u[2]; v16bf v; };
static_assert(sizeof(Frag16) == 32, "frag size");
union U2 { unsigned int u; bf16_t h[2]; };

static __device__ __forceinline__ v8f wmma_bf16f32(v16bf a, v16bf b, v8f c) {
  // (neg_a, A, neg_b, B, c_mod, C, reuse_a, reuse_b)
  return __builtin_amdgcn_wmma_f32_16x16x32_bf16(false, a, false, b, (short)0, c,
                                                 false, false);
}

// 16B global -> LDS async copy (per-lane); no-op fallback when absent.
static __device__ __forceinline__ void async_copy_b128(const bf16_t* g, bf16_t* l) {
#if HAVE_ASYNC_LDS
  __builtin_amdgcn_global_load_async_to_lds_b128((as1_v4i)g, (as3_v4i)l, 0, 0);
#else
  (void)g; (void)l;
#endif
}

static __device__ __forceinline__ void async_wait0() {
#if HAVE_ASYNC_LDS
#  if defined(__has_builtin) && __has_builtin(__builtin_amdgcn_s_wait_asynccnt)
  __builtin_amdgcn_s_wait_asynccnt(0);
#  else
  asm volatile("s_wait_asynccnt 0x0" ::: "memory");
#  endif
#endif
}

// A fragment 16x32 (MxK), row-major source, stride lda elements (lda%8==0).
// ISA: lane m=L%16, h=L/16; elems 0..7 -> K=h*8+i, elems 8..15 -> K=16+h*8+i.
static __device__ __forceinline__ v16bf load_a_frag(const bf16_t* base, int lda, int lane) {
  const int m = lane & 15, h = lane >> 4;
  const bf16_t* rp = base + (size_t)m * (size_t)lda;
  Frag16 f;
  f.u[0] = *(const u32x4*)(rp + h * 8);
  f.u[1] = *(const u32x4*)(rp + 16 + h * 8);
  return f.v;
}

// B fragment 32x16 (KxN) from TRANSPOSED storage baseT[n][k], stride ldt.
// ISA: lane n=L%16, h=L/16; elems 0..7 -> K=h*8+i, elems 8..15 -> K=16+h*8+i.
static __device__ __forceinline__ v16bf load_b_frag_t(const bf16_t* baseT, int ldt, int lane) {
  const int n = lane & 15, h = lane >> 4;
  const bf16_t* rp = baseT + (size_t)n * (size_t)ldt + h * 8;
  Frag16 f;
  f.u[0] = *(const u32x4*)(rp);
  f.u[1] = *(const u32x4*)(rp + 16);
  return f.v;
}

// ---------------------------------------------------------------------------
// fp32 -> bf16 conversion (vectorized, n % 1024 == 0)
// ---------------------------------------------------------------------------
__global__ __launch_bounds__(256) void f32_to_bf16_kernel(const float* __restrict__ in,
                                                          bf16_t* __restrict__ out) {
  size_t i = ((size_t)blockIdx.x * 256 + threadIdx.x) * 4;
  float4 f = *(const float4*)(in + i);
  union { bf16_t h[4]; unsigned long long u; } o;
  o.h[0] = (bf16_t)f.x; o.h[1] = (bf16_t)f.y;
  o.h[2] = (bf16_t)f.z; o.h[3] = (bf16_t)f.w;
  *(unsigned long long*)(out + i) = o.u;
}

// ---------------------------------------------------------------------------
// Tiled bf16 GEMM: C[M,N] = A[M,K] x B[K,N], f32 accumulate.
// Block tile 128x128, BK=32, 256 threads = 8 wave32 (4x2 grid, wave tile 32x64).
// Ping-pong LDS; A tile via async global->LDS DMA, B tile via staged transpose.
// One barrier per K-step. M,N % 128 == 0, K % 32 == 0.
// ---------------------------------------------------------------------------
template <typename OutT>
__global__ __launch_bounds__(256) void gemm_bf16_kernel(const bf16_t* __restrict__ A,
                                                        const bf16_t* __restrict__ B,
                                                        OutT* __restrict__ C,
                                                        int M, int N, int K) {
  constexpr int LDA = 40;  // 32 + 8 pad (16B-aligned rows)
  __shared__ __align__(16) bf16_t sA[2][128 * LDA];   // [m][k]
  __shared__ __align__(16) bf16_t sBt[2][128 * LDA];  // [n][k] (transposed)

  const int t = threadIdx.x;
  const int lane = t & 31, wv = t >> 5;
  const int wm = wv >> 1, wn = wv & 1;  // 4x2 waves
  const int bm = blockIdx.y * 128, bn = blockIdx.x * 128;

  // ---- per-thread staging addresses (computed once, stride-advanced) ----
  const int rowA = t >> 2, c8A = t & 3;  // A: 2 chunks of 64 rows
  const bf16_t* pA0 = A + (size_t)(bm + rowA) * K + c8A * 8;
  const bf16_t* pA1 = pA0 + (size_t)64 * K;
  const int offA0 = rowA * LDA + c8A * 8;
  const int offA1 = (rowA + 64) * LDA + c8A * 8;

  const int kkB = t >> 6, n2B = t & 63;  // B: 8 chunks of 4 k-rows
  const bf16_t* pB = B + (size_t)kkB * N + bn + n2B * 2;
  const int offB0 = (n2B * 2) * LDA + kkB;
  const int offB1 = (n2B * 2 + 1) * LDA + kkB;

  unsigned int bR[8];
#if !HAVE_ASYNC_LDS
  u32x4 aR[2];
#endif

  // ---- preload tile 0 ----
#if HAVE_ASYNC_LDS
  async_copy_b128(pA0, &sA[0][offA0]);
  async_copy_b128(pA1, &sA[0][offA1]);
#else
  aR[0] = *(const u32x4*)pA0;
  aR[1] = *(const u32x4*)pA1;
#endif
#pragma unroll
  for (int i = 0; i < 8; ++i)
    bR[i] = *(const unsigned int*)(pB + (size_t)(i * 4) * N);

  v8f acc[2][4] = {};

  int p = 0;
  for (int k0 = 0; k0 < K; k0 += 32, p ^= 1) {
    // ---- commit staged B (and A if no DMA) into buffer p ----
#if !HAVE_ASYNC_LDS
    *(u32x4*)&sA[p][offA0] = aR[0];
    *(u32x4*)&sA[p][offA1] = aR[1];
#endif
#pragma unroll
    for (int i = 0; i < 8; ++i) {
      U2 w; w.u = bR[i];
      sBt[p][offB0 + i * 4] = w.h[0];
      sBt[p][offB1 + i * 4] = w.h[1];
    }
    async_wait0();     // our async writes into buffer p have landed
    __syncthreads();   // everyone's writes visible; buffer p^1 free

    // ---- issue next tile into buffer p^1 (hidden behind WMMA below) ----
    if (k0 + 32 < K) {
      pA0 += 32; pA1 += 32; pB += (size_t)32 * N;
#if HAVE_ASYNC_LDS
      async_copy_b128(pA0, &sA[p ^ 1][offA0]);
      async_copy_b128(pA1, &sA[p ^ 1][offA1]);
#else
      aR[0] = *(const u32x4*)pA0;
      aR[1] = *(const u32x4*)pA1;
#endif
#pragma unroll
      for (int i = 0; i < 8; ++i)
        bR[i] = *(const unsigned int*)(pB + (size_t)(i * 4) * N);
    }

    // ---- compute from buffer p ----
    v16bf af[2], bfr[4];
#pragma unroll
    for (int i = 0; i < 2; ++i)
      af[i] = load_a_frag(&sA[p][(wm * 32 + i * 16) * LDA], LDA, lane);
#pragma unroll
    for (int j = 0; j < 4; ++j)
      bfr[j] = load_b_frag_t(&sBt[p][(wn * 64 + j * 16) * LDA], LDA, lane);
#pragma unroll
    for (int i = 0; i < 2; ++i)
#pragma unroll
      for (int j = 0; j < 4; ++j)
        acc[i][j] = wmma_bf16f32(af[i], bfr[j], acc[i][j]);
  }

  // C/D layout: lane n=L%16, half h=L/16; vgpr r -> row h*8+r.
  const int h = lane >> 4, n = lane & 15;
#pragma unroll
  for (int i = 0; i < 2; ++i)
#pragma unroll
    for (int j = 0; j < 4; ++j)
#pragma unroll
      for (int r = 0; r < 8; ++r) {
        int row = bm + wm * 32 + i * 16 + h * 8 + r;
        int col = bn + wn * 64 + j * 16 + n;
        C[(size_t)row * N + col] = (OutT)acc[i][j][r];
      }
}

// ---------------------------------------------------------------------------
// Flash attention (causal). One block = 128 query rows of one (batch, head).
// 8 waves; wave wv owns rows [qbase+wv*16, +16). Q frags live in VGPRs.
// K tiles async-DMA'd to LDS, V tiles staged+transposed; ping-pong buffers,
// one barrier per key tile. Online softmax in log2 domain; causal mask only
// on diagonal-crossing tiles.
// qkv layout: [B*S, 6144], Q at +0, K at +2048, V at +4096, head*128 offset.
// ---------------------------------------------------------------------------
__global__ __launch_bounds__(256) void flash_attn_kernel(const bf16_t* __restrict__ qkv,
                                                         bf16_t* __restrict__ o) {
  constexpr int S = 2048, D = 2048, H = 16, HD = 128, QKVW = 6144;
  constexpr int LDV = 40;  // 32 keys + 8 pad
  __shared__ __align__(16) bf16_t sK[2][32 * 128];    // [key][hd]
  __shared__ __align__(16) bf16_t sVt[2][128 * LDV];  // [hd][key]
  __shared__ __align__(16) bf16_t sP[8 * 16 * 32];    // per-wave P tile [row][key]

  const int t = threadIdx.x, lane = t & 31, wv = t >> 5;
  const int h = lane >> 4, nn = lane & 15;
  const int qt = blockIdx.x;  // query tile (16 of them)
  const int bh = blockIdx.y;  // 0..31
  const int b = bh / H, head = bh % H;
  const int qbase = qt * 128;
  const size_t rowBase = (size_t)b * S;
  // softmax scale folded with log2(e): scores scaled once, exp2f afterwards.
  const float kScl2 = 0.08838834764831845f * 1.4426950408889634f;

  // ---- Q fragments (4 x 16x32 across HD=128), straight from global ----
  v16bf qf[4];
  {
    const bf16_t* qptr = qkv + (rowBase + qbase + wv * 16) * QKVW + head * HD;
#pragma unroll
    for (int kk = 0; kk < 4; ++kk) qf[kk] = load_a_frag(qptr + kk * 32, QKVW, lane);
  }

  // ---- per-thread K/V staging addresses ----
  const int krK = t >> 4, c8K = t & 15;  // K: 2 chunks of 16 rows
  const bf16_t* pK = qkv + (rowBase + krK) * QKVW + D + head * HD + c8K * 8;
  const int offK = krK * 128 + c8K * 8;

  const int krV = t >> 6, c2V = t & 63;  // V: 8 chunks of 4 rows (transposed)
  const bf16_t* pV = qkv + (rowBase + krV) * QKVW + 2 * D + head * HD + c2V * 2;
  const int offV0 = (c2V * 2) * LDV + krV;
  const int offV1 = (c2V * 2 + 1) * LDV + krV;

  unsigned int vR[8];
#if !HAVE_ASYNC_LDS
  u32x4 kR[2];
#endif

  // ---- preload tile 0 ----
#if HAVE_ASYNC_LDS
  async_copy_b128(pK, &sK[0][offK]);
  async_copy_b128(pK + (size_t)16 * QKVW, &sK[0][offK + 16 * 128]);
#else
  kR[0] = *(const u32x4*)pK;
  kR[1] = *(const u32x4*)(pK + (size_t)16 * QKVW);
#endif
#pragma unroll
  for (int i = 0; i < 8; ++i)
    vR[i] = *(const unsigned int*)(pV + (size_t)(i * 4) * QKVW);

  v8f acc[8] = {};
  float m_run[8], l_run[8];
#pragma unroll
  for (int r = 0; r < 8; ++r) { m_run[r] = -1e30f; l_run[r] = 0.0f; }

  const int qmin = qbase + wv * 16;  // wave's lowest query row
  const int ktmax = (qt + 1) * 4;    // causal: only key tiles <= query tile
  int p = 0;
  for (int kt = 0; kt < ktmax; ++kt, p ^= 1) {
    // ---- commit staged V (and K if no DMA) into buffer p ----
#if !HAVE_ASYNC_LDS
    *(u32x4*)&sK[p][offK] = kR[0];
    *(u32x4*)&sK[p][offK + 16 * 128] = kR[1];
#endif
#pragma unroll
    for (int i = 0; i < 8; ++i) {
      U2 w; w.u = vR[i];
      sVt[p][offV0 + i * 4] = w.h[0];
      sVt[p][offV1 + i * 4] = w.h[1];
    }
    async_wait0();
    __syncthreads();

    // ---- issue next tile into buffer p^1 (hidden behind WMMA below) ----
    if (kt + 1 < ktmax) {
      pK += (size_t)32 * QKVW;
      pV += (size_t)32 * QKVW;
#if HAVE_ASYNC_LDS
      async_copy_b128(pK, &sK[p ^ 1][offK]);
      async_copy_b128(pK + (size_t)16 * QKVW, &sK[p ^ 1][offK + 16 * 128]);
#else
      kR[0] = *(const u32x4*)pK;
      kR[1] = *(const u32x4*)(pK + (size_t)16 * QKVW);
#endif
#pragma unroll
      for (int i = 0; i < 8; ++i)
        vR[i] = *(const unsigned int*)(pV + (size_t)(i * 4) * QKVW);
    }

    // ---- scores: S[16 rows][32 keys] = Q x K^T (contraction over HD) ----
    v8f s0 = {}, s1 = {};
#pragma unroll
    for (int kk = 0; kk < 4; ++kk) {
      v16bf b0 = load_b_frag_t(&sK[p][0 * 128 + kk * 32], 128, lane);   // keys 0..15
      v16bf b1 = load_b_frag_t(&sK[p][16 * 128 + kk * 32], 128, lane);  // keys 16..31
      s0 = wmma_bf16f32(qf[kk], b0, s0);
      s1 = wmma_bf16f32(qf[kk], b1, s1);
    }

    // ---- scale (log2 domain); causal mask only on diagonal-crossing tiles ----
    const int qrow0 = qbase + wv * 16 + h * 8;
    const int key0 = kt * 32 + nn;
    float p0[8], p1[8];
#pragma unroll
    for (int r = 0; r < 8; ++r) { p0[r] = s0[r] * kScl2; p1[r] = s1[r] * kScl2; }
    if (kt * 32 + 31 > qmin) {  // wave-uniform -> scalar branch, EXEC untouched
#pragma unroll
      for (int r = 0; r < 8; ++r) {
        int q = qrow0 + r;
        if (key0 > q)      p0[r] = -1e30f;
        if (key0 + 16 > q) p1[r] = -1e30f;
      }
    }

    // ---- online softmax (row stats live in the 16-lane half) ----
#pragma unroll
    for (int r = 0; r < 8; ++r) {
      float mx = fmaxf(p0[r], p1[r]);
      mx = fmaxf(mx, __shfl_xor(mx, 1, 32));
      mx = fmaxf(mx, __shfl_xor(mx, 2, 32));
      mx = fmaxf(mx, __shfl_xor(mx, 4, 32));
      mx = fmaxf(mx, __shfl_xor(mx, 8, 32));
      float mnew = fmaxf(m_run[r], mx);
      float scale = exp2f(m_run[r] - mnew);
      float e0 = exp2f(p0[r] - mnew);
      float e1 = exp2f(p1[r] - mnew);
      float sum = e0 + e1;
      sum += __shfl_xor(sum, 1, 32);
      sum += __shfl_xor(sum, 2, 32);
      sum += __shfl_xor(sum, 4, 32);
      sum += __shfl_xor(sum, 8, 32);
      l_run[r] = l_run[r] * scale + sum;
      m_run[r] = mnew;
#pragma unroll
      for (int f = 0; f < 8; ++f) acc[f][r] *= scale;
      p0[r] = e0; p1[r] = e1;
    }

    // ---- reshape P: C-layout -> A-layout via per-wave LDS tile ----
    bf16_t* pw = &sP[wv * 16 * 32];
#pragma unroll
    for (int r = 0; r < 8; ++r) {
      pw[(h * 8 + r) * 32 + nn]      = (bf16_t)p0[r];
      pw[(h * 8 + r) * 32 + 16 + nn] = (bf16_t)p1[r];
    }
    v16bf pf = load_a_frag(pw, 32, lane);

    // ---- O += P x V (contraction over 32 keys, 8 hd-column frags) ----
#pragma unroll
    for (int f = 0; f < 8; ++f) {
      v16bf vf = load_b_frag_t(&sVt[p][(f * 16) * LDV], LDV, lane);
      acc[f] = wmma_bf16f32(pf, vf, acc[f]);
    }
  }

  // ---- normalize, convert, store bf16 O in [B*S, D] layout ----
#pragma unroll
  for (int r = 0; r < 8; ++r) {
    float inv = 1.0f / l_run[r];
    int q = qbase + wv * 16 + h * 8 + r;
#pragma unroll
    for (int f = 0; f < 8; ++f) {
      float val = acc[f][r] * inv;
      o[(rowBase + q) * D + head * HD + f * 16 + nn] = (bf16_t)val;
    }
  }
}

// ---------------------------------------------------------------------------
// Host launch
// ---------------------------------------------------------------------------
extern "C" void kernel_launch(void* const* d_in, const int* in_sizes, int n_in,
                              void* d_out, int out_size, void* d_ws, size_t ws_size,
                              hipStream_t stream) {
  (void)in_sizes; (void)n_in; (void)out_size; (void)ws_size;
  constexpr int Dm = 2048, Hh = 16, S = 2048, Bb = 2;
  constexpr int Mrows = Bb * S;  // 4096
  constexpr int NQKV = 3 * Dm;   // 6144

  const float* x    = (const float*)d_in[0];  // [2,2048,2048]
  const float* Wqkv = (const float*)d_in[1];  // [2048,6144]
  const float* Wo   = (const float*)d_in[2];  // [2048,2048]
  float* out = (float*)d_out;                 // [2,2048,2048]

  constexpr size_t NX    = (size_t)Mrows * Dm;    // 8388608
  constexpr size_t NWQKV = (size_t)Dm * NQKV;     // 12582912
  constexpr size_t NWO   = (size_t)Dm * Dm;       // 4194304
  constexpr size_t NQKVO = (size_t)Mrows * NQKV;  // 25165824

  bf16_t* xb    = (bf16_t*)d_ws;
  bf16_t* wqkvb = xb + NX;
  bf16_t* wob   = wqkvb + NWQKV;
  bf16_t* qkvb  = wob + NWO;
  bf16_t* ob    = qkvb + NQKVO;

  // 1) fp32 -> bf16 converts
  f32_to_bf16_kernel<<<NX / 1024, 256, 0, stream>>>(x, xb);
  f32_to_bf16_kernel<<<NWQKV / 1024, 256, 0, stream>>>(Wqkv, wqkvb);
  f32_to_bf16_kernel<<<NWO / 1024, 256, 0, stream>>>(Wo, wob);

  // 2) qkv = x @ Wqkv  (bf16 out)
  gemm_bf16_kernel<bf16_t><<<dim3(NQKV / 128, Mrows / 128), 256, 0, stream>>>(
      xb, wqkvb, qkvb, Mrows, NQKV, Dm);

  // 3) flash attention -> bf16 o [B*S, D]
  flash_attn_kernel<<<dim3(S / 128, Bb * Hh), 256, 0, stream>>>(qkvb, ob);

  // 4) out = o @ Wo  (fp32 out)
  gemm_bf16_kernel<float><<<dim3(Dm / 128, Mrows / 128), 256, 0, stream>>>(
      ob, wob, out, Mrows, Dm, Dm);
}
